// Head_38998303048530
// MI455X (gfx1250) — compile-verified, matrix operand order checked by reading, not run on hardware
//
#include <hip/hip_runtime.h>

// Shapes from reference: B=8, T=2048, HID=1024, HS=64
#define B_   8
#define T_   2048
#define HID_ 1024
#define HS_  64

typedef __attribute__((ext_vector_type(16))) _Float16 v16h;
typedef __attribute__((ext_vector_type(8)))  _Float16 v8h;
typedef __attribute__((ext_vector_type(8)))  float    v8f;
typedef __attribute__((ext_vector_type(4)))  float    v4f;

union Frag { v16h v; v8h h[2]; };

static __device__ __forceinline__ v8f wmma16x16x32(v16h a, v16h b, v8f c) {
  // 8 args: (neg_a, A, neg_b, B, c_mod, C, reuse_a, reuse_b)
  return __builtin_amdgcn_wmma_f32_16x16x32_f16(false, a, false, b, (short)0, c,
                                                false, false);
}

static __device__ __forceinline__ v8f vzero8() {
  v8f z;
  #pragma unroll
  for (int i = 0; i < 8; ++i) z[i] = 0.0f;
  return z;
}

// ---------------------------------------------------------------------------
// Kernel 0: transpose + fp32->f16 convert the three weights.
// Wt layout: [wi][n (0..63)][k (0..1023)] f16   (N-major so B-frags are
// contiguous runs of 8 halves per lane)
// ---------------------------------------------------------------------------
__global__ void __launch_bounds__(256)
wt_transpose_kernel(const float* __restrict__ Wq, const float* __restrict__ Wk,
                    const float* __restrict__ Wv, _Float16* __restrict__ Wt) {
  int idx = blockIdx.x * 256 + threadIdx.x;   // 3*64*1024 = 196608 total
  int wi  = idx >> 16;
  int rem = idx & 65535;
  int n   = rem >> 10;
  int k   = rem & 1023;
  const float* W = (wi == 0) ? Wq : (wi == 1) ? Wk : Wv;
  Wt[idx] = (_Float16)W[k * HS_ + n];
}

// ---------------------------------------------------------------------------
// Kernel 1: QKV projection.  One block = 128 rows of x, 8 waves, wave w owns
// rows [blk*128 + w*16, +16).  Each wave keeps 12 fp32 accumulator tiles
// (q,k,v) x (4 N-blocks of 16).  K loop step 32 (WMMA f16 K-dim).
// Outputs: Qh [B*T,64] f16, Kh [B*T,64] f16, VhT [B,64,T] f16 (transposed).
// ---------------------------------------------------------------------------
__global__ void __launch_bounds__(256)
qkv_kernel(const float* __restrict__ x, const _Float16* __restrict__ Wt,
           _Float16* __restrict__ Qh, _Float16* __restrict__ Kh,
           _Float16* __restrict__ VhT) {
  const int lane = threadIdx.x & 31;
  const int wv   = threadIdx.x >> 5;   // 0..7
  const int l16  = lane & 15;
  const int hlf  = lane >> 4;          // 0/1 : which 16-lane half
  const int kb   = hlf * 8;            // per-lane K base inside 16-wide half
  const long rbase = (long)blockIdx.x * 128 + wv * 16;
  const long arow  = rbase + l16;      // A-matrix row for this lane

  v8f acc[12];
  #pragma unroll
  for (int i = 0; i < 12; ++i) acc[i] = vzero8();

  for (int k0 = 0; k0 < HID_; k0 += 32) {
    // ---- A fragment: x rows, fp32 -> f16.  Layout per ISA 16-bit A:
    // low 8 halves: K = k0+kb+0..7 ; high 8 halves: K = k0+16+kb+0..7
    const float* xp = x + arow * HID_ + k0 + kb;
    v4f f0 = *(const v4f*)(xp + 0);
    v4f f1 = *(const v4f*)(xp + 4);
    v4f f2 = *(const v4f*)(xp + 16);
    v4f f3 = *(const v4f*)(xp + 20);
    Frag a;
    #pragma unroll
    for (int i = 0; i < 4; ++i) {
      a.v[i]      = (_Float16)f0[i];
      a.v[4 + i]  = (_Float16)f1[i];
      a.v[8 + i]  = (_Float16)f2[i];
      a.v[12 + i] = (_Float16)f3[i];
    }
    #pragma unroll
    for (int wi = 0; wi < 3; ++wi) {
      #pragma unroll
      for (int nb = 0; nb < 4; ++nb) {
        // B fragment: lane holds column n = nb*16+l16, contiguous in K
        const _Float16* wp = Wt + wi * (HS_ * HID_) +
                             (nb * 16 + l16) * HID_ + k0 + kb;
        Frag bm;
        bm.h[0] = *(const v8h*)(wp);
        bm.h[1] = *(const v8h*)(wp + 16);
        acc[wi * 4 + nb] = wmma16x16x32(a.v, bm.v, acc[wi * 4 + nb]);
      }
    }
  }

  // ---- store.  D-layout: element (M,N): vgpr r -> M = r + 8*hlf, N = l16.
  #pragma unroll
  for (int wi = 0; wi < 3; ++wi) {
    #pragma unroll
    for (int nb = 0; nb < 4; ++nb) {
      #pragma unroll
      for (int r = 0; r < 8; ++r) {
        long row = rbase + r + 8 * hlf;          // global (b*T + t)
        int  n   = nb * 16 + l16;
        _Float16 hv = (_Float16)acc[wi * 4 + nb][r];
        if (wi == 0) {
          Qh[row * HS_ + n] = hv;
        } else if (wi == 1) {
          Kh[row * HS_ + n] = hv;
        } else {
          long bz = row >> 11;                   // /T_
          long t  = row & (T_ - 1);
          VhT[(bz * HS_ + n) * T_ + t] = hv;     // transposed V
        }
      }
    }
  }
}

// ---------------------------------------------------------------------------
// Kernel 2: causal flash attention over f16 Q,K,Vt with fp32 accumulation.
// Block = (batch, 128-query tile); wave w owns 16 queries.  Key blocks of 128.
// ---------------------------------------------------------------------------
__global__ void __launch_bounds__(256)
flash_kernel(const _Float16* __restrict__ Qh, const _Float16* __restrict__ Kh,
             const _Float16* __restrict__ VhT, float* __restrict__ out) {
  __shared__ alignas(16) _Float16 Pbuf[8][16 * 128];  // per-wave P staging

  const int lane = threadIdx.x & 31;
  const int wv   = threadIdx.x >> 5;
  const int l16  = lane & 15;
  const int hlf  = lane >> 4;
  const int kb   = hlf * 8;
  const int bz   = blockIdx.x >> 4;        // batch
  const int qt   = blockIdx.x & 15;        // query tile
  const int qbase = qt * 128;
  const int qr0   = qbase + wv * 16;       // this wave's first query row

  // Q fragments (A-matrix), hs chunks c=0,1 — kept in registers whole loop
  Frag aq[2];
  {
    const _Float16* qp = Qh + ((long)(bz * T_ + qr0 + l16)) * HS_ + kb;
    #pragma unroll
    for (int c = 0; c < 2; ++c) {
      aq[c].h[0] = *(const v8h*)(qp + c * 32);
      aq[c].h[1] = *(const v8h*)(qp + c * 32 + 16);
    }
  }

  v8f o[4];
  #pragma unroll
  for (int i = 0; i < 4; ++i) o[i] = vzero8();
  float mrow[8], lrow[8];
  #pragma unroll
  for (int r = 0; r < 8; ++r) { mrow[r] = -1e30f; lrow[r] = 0.0f; }

  const float sc = 0.03125f;               // 1024^-0.5 (hidden-dim scaling)
  const int nkt = qt + 1;                  // causal: key tiles up to diagonal

  for (int kt = 0; kt < nkt; ++kt) {
    const int kbase = kt * 128;
    if (kt + 1 < nkt) {                    // prefetch next K tile
      __builtin_prefetch(Kh + ((long)(bz * T_ + kbase + 128)) * HS_, 0, 1);
    }

    // ---- S = Q * K^T  (16 WMMAs) ----
    v8f s[8];
    #pragma unroll
    for (int nb = 0; nb < 8; ++nb) s[nb] = vzero8();
    #pragma unroll
    for (int nb = 0; nb < 8; ++nb) {
      const _Float16* kp =
          Kh + ((long)(bz * T_ + kbase + nb * 16 + l16)) * HS_ + kb;
      #pragma unroll
      for (int c = 0; c < 2; ++c) {
        Frag bk;
        bk.h[0] = *(const v8h*)(kp + c * 32);
        bk.h[1] = *(const v8h*)(kp + c * 32 + 16);
        s[nb] = wmma16x16x32(aq[c].v, bk.v, s[nb]);
      }
    }

    // ---- scale + causal mask ----
    #pragma unroll
    for (int nb = 0; nb < 8; ++nb) {
      int key = kbase + nb * 16 + l16;
      #pragma unroll
      for (int r = 0; r < 8; ++r) {
        int qg = qr0 + r + 8 * hlf;
        float v = s[nb][r] * sc;
        s[nb][r] = (key > qg) ? -1e30f : v;
      }
    }

    // ---- online softmax per row; write P (f16) to LDS in A-layout order ---
    #pragma unroll
    for (int r = 0; r < 8; ++r) {
      float mx = -1e30f;
      #pragma unroll
      for (int nb = 0; nb < 8; ++nb) mx = fmaxf(mx, s[nb][r]);
      #pragma unroll
      for (int off = 1; off < 16; off <<= 1)
        mx = fmaxf(mx, __shfl_xor(mx, off, 32));   // reduce within 16-lane half
      float mnew = fmaxf(mrow[r], mx);
      float corr = __expf(mrow[r] - mnew);
      float rs = 0.0f;
      int prow = (r + 8 * hlf) * 128;
      #pragma unroll
      for (int nb = 0; nb < 8; ++nb) {
        float p = __expf(s[nb][r] - mnew);
        rs += p;
        Pbuf[wv][prow + nb * 16 + l16] = (_Float16)p;
      }
      #pragma unroll
      for (int off = 1; off < 16; off <<= 1) rs += __shfl_xor(rs, off, 32);
      lrow[r] = lrow[r] * corr + rs;
      mrow[r] = mnew;
      #pragma unroll
      for (int n4 = 0; n4 < 4; ++n4) o[n4][r] *= corr;
    }

    __syncthreads();   // make P visible across lanes (and order DS ops)

    // ---- O += P * V  (16 WMMAs; V^T gives contiguous B-frag loads) ----
    #pragma unroll
    for (int c = 0; c < 4; ++c) {           // key chunks of 32
      Frag ap;
      const _Float16* pp = &Pbuf[wv][l16 * 128 + c * 32 + kb];
      ap.h[0] = *(const v8h*)(pp);
      ap.h[1] = *(const v8h*)(pp + 16);
      #pragma unroll
      for (int nb = 0; nb < 4; ++nb) {      // hs blocks of 16
        const _Float16* vp =
            VhT + ((long)(bz * HS_ + nb * 16 + l16)) * T_ + kbase + c * 32 + kb;
        Frag bv;
        bv.h[0] = *(const v8h*)(vp);
        bv.h[1] = *(const v8h*)(vp + 16);
        o[nb] = wmma16x16x32(ap.v, bv.v, o[nb]);
      }
    }

    __syncthreads();
  }

  // ---- epilogue: O / l, fp32 store ----
  #pragma unroll
  for (int r = 0; r < 8; ++r) {
    int   tq  = qr0 + r + 8 * hlf;
    float inv = 1.0f / lrow[r];
    #pragma unroll
    for (int nb = 0; nb < 4; ++nb) {
      out[((long)(bz * T_ + tq)) * HS_ + nb * 16 + l16] = o[nb][r] * inv;
    }
  }
}

// ---------------------------------------------------------------------------
extern "C" void kernel_launch(void* const* d_in, const int* in_sizes, int n_in,
                              void* d_out, int out_size, void* d_ws,
                              size_t ws_size, hipStream_t stream) {
  const float* x  = (const float*)d_in[0];
  const float* Wq = (const float*)d_in[1];
  const float* Wk = (const float*)d_in[2];
  const float* Wv = (const float*)d_in[3];
  float* out = (float*)d_out;

  // Workspace layout (f16): Wt[3*64*1024] | Qh[B*T*64] | Kh[...] | VhT[...]
  _Float16* Wt  = (_Float16*)d_ws;
  _Float16* Qh  = Wt + (long)3 * HS_ * HID_;
  _Float16* Kh  = Qh + (long)B_ * T_ * HS_;
  _Float16* VhT = Kh + (long)B_ * T_ * HS_;

  wt_transpose_kernel<<<(3 * HS_ * HID_) / 256, 256, 0, stream>>>(Wq, Wk, Wv, Wt);
  qkv_kernel<<<(B_ * T_) / 128, 256, 0, stream>>>(x, Wt, Qh, Kh, VhT);
  flash_kernel<<<B_ * (T_ / 128), 256, 0, stream>>>(Qh, Kh, VhT, out);
}
